// BatchTripletMarginLoss_6038724018541
// MI455X (gfx1250) — matrix operand order repaired and checked by reading, not compiled, not run on hardware
//
#include <hip/hip_runtime.h>

typedef _Float16 half_t;
typedef _Float16 v16h __attribute__((ext_vector_type(16)));
typedef _Float16 v8h  __attribute__((ext_vector_type(8)));
typedef float    v8f  __attribute__((ext_vector_type(8)));

#define BSZ 512
#define DIM 768
#define MARGIN 1.0f

// ---------------------------------------------------------------------------
// Kernel 1: f32 -> f16 copy of embeddings, exact f32 row sq-norms,
//           and zero the global accumulators (stream-ordered, graph-safe).
// ---------------------------------------------------------------------------
__global__ __launch_bounds__(256) void prep_kernel(
    const float* __restrict__ emb,
    half_t* __restrict__ e16,
    float* __restrict__ sqn,
    double* __restrict__ acc) {
  __shared__ float red[256];
  const int row = blockIdx.x;
  const int t   = threadIdx.x;
  if (row == 0 && t == 0) { acc[0] = 0.0; acc[1] = 0.0; }

  const float* r = emb + (size_t)row * DIM;
  half_t*      w = e16 + (size_t)row * DIM;
  float s = 0.0f;
  for (int k = t; k < DIM; k += 256) {
    float v = r[k];
    s += v * v;
    w[k] = (half_t)v;
  }
  red[t] = s;
  __syncthreads();
  for (int off = 128; off > 0; off >>= 1) {
    if (t < off) red[t] += red[t + off];
    __syncthreads();
  }
  if (t == 0) sqn[row] = red[0];
}

// ---------------------------------------------------------------------------
// Kernel 2: G = E_f16 * E_f16^T via v_wmma_f32_16x16x32_f16, fused into
//           dist[i][j] = sqrt(max(sqn[i] + sqn[j] - 2*G[i][j], 0)).
// One wave per 16x16 output tile; 32x32 tiles; 4 waves/block, 256 blocks.
// ---------------------------------------------------------------------------
__global__ __launch_bounds__(128) void gemm_dist_kernel(
    const half_t* __restrict__ e16,
    const float* __restrict__ sqn,
    float* __restrict__ dist) {
  const int wave   = blockIdx.x * 4 + (threadIdx.x >> 5);
  const int lane   = threadIdx.x & 31;
  const int tm     = (wave >> 5) << 4;   // tile row base
  const int tn     = (wave & 31) << 4;   // tile col base
  const int laneLo = lane & 15;
  const int laneHi = lane >> 4;

  // A fragment: row tm+laneLo, K-chunk at laneHi*16 .. +15
  const half_t* arow = e16 + (size_t)(tm + laneLo) * DIM + laneHi * 16;
  // B fragment: row tn+laneLo (column of B), K-chunk at laneHi*16 .. +15
  const half_t* brow = e16 + (size_t)(tn + laneLo) * DIM + laneHi * 16;

  v8f c = {0.f, 0.f, 0.f, 0.f, 0.f, 0.f, 0.f, 0.f};

  for (int k0 = 0; k0 < DIM; k0 += 32) {
    v8h a0 = *(const v8h*)(arow + k0);
    v8h a1 = *(const v8h*)(arow + k0 + 8);
    v8h b0 = *(const v8h*)(brow + k0);
    v8h b1 = *(const v8h*)(brow + k0 + 8);
    v16h a, b;
#pragma unroll
    for (int i = 0; i < 8; ++i) {
      a[i] = a0[i]; a[i + 8] = a1[i];
      b[i] = b0[i]; b[i + 8] = b1[i];
    }
    // (neg_a, A, neg_b, B, c_mod, C, reuse_a, reuse_b)
    c = __builtin_amdgcn_wmma_f32_16x16x32_f16(false, a, false, b,
                                               (short)0, c, false, false);
  }

  // Epilogue: C/D layout -> VGPR r holds (row tm+r+8*laneHi, col tn+laneLo)
  const int colN   = tn + laneLo;
  const float snC  = sqn[colN];
#pragma unroll
  for (int r = 0; r < 8; ++r) {
    const int rowM = tm + r + laneHi * 8;
    float sq = sqn[rowM] + snC - 2.0f * c[r];
    float d  = sq > 0.0f ? sqrtf(sq) : 0.0f;
    dist[(size_t)rowM * BSZ + colN] = d;
  }
}

// ---------------------------------------------------------------------------
// Kernel 3: per-anchor brute-force triplet sum (faithful semantics: every
// invalid triplet contributes relu(margin) = 1). Block = one anchor.
// ---------------------------------------------------------------------------
__global__ __launch_bounds__(256) void triplet_kernel(
    const int* __restrict__ types,
    const float* __restrict__ dist,
    double* __restrict__ acc) {
  __shared__ float dsh[BSZ];
  __shared__ int   tsh[BSZ];
  __shared__ float redf[256];
  __shared__ int   redi[256];

  const int a = blockIdx.x;
  const int t = threadIdx.x;
  const float* drow = dist + (size_t)a * BSZ;
  for (int j = t; j < BSZ; j += 256) { dsh[j] = drow[j]; tsh[j] = types[j]; }
  __syncthreads();

  const int ta = tsh[a];
  float local = 0.0f;
  int   cnt   = 0;
  for (int p = t; p < BSZ; p += 256) {
    const bool cp = (tsh[p] == ta);
    if (!cp) { local += (float)BSZ; continue; }  // all n invalid -> +1 each
    ++cnt;
    const float dap = dsh[p];
    for (int n = 0; n < BSZ; ++n) {
      const bool valid = (tsh[n] != ta);
      local += valid ? fmaxf(dap - dsh[n] + MARGIN, 0.0f) : 1.0f;
    }
  }
  redf[t] = local;
  redi[t] = cnt;
  __syncthreads();
  for (int off = 128; off > 0; off >>= 1) {
    if (t < off) { redf[t] += redf[t + off]; redi[t] += redi[t + off]; }
    __syncthreads();
  }
  if (t == 0) {
    atomicAdd(&acc[0], (double)redf[0]);
    const int cp = redi[0];  // |P_a| including a itself
    atomicAdd(&acc[1], (double)cp * (double)(BSZ - cp));
  }
}

// ---------------------------------------------------------------------------
// Kernel 4: final divide.
// ---------------------------------------------------------------------------
__global__ void finalize_kernel(const double* __restrict__ acc,
                                float* __restrict__ out) {
  out[0] = (float)(acc[0] / acc[1]);
}

extern "C" void kernel_launch(void* const* d_in, const int* in_sizes, int n_in,
                              void* d_out, int out_size, void* d_ws, size_t ws_size,
                              hipStream_t stream) {
  const int*   types = (const int*)d_in[0];    // entity_types (int)
  const float* emb   = (const float*)d_in[1];  // embeddings f32 [512,768]

  char* ws = (char*)d_ws;
  half_t* e16 = (half_t*)ws;                                   // 786,432 B
  float*  dst = (float*)(ws + 786432);                         // 1,048,576 B
  float*  sqn = (float*)(ws + 786432 + 1048576);               // 2,048 B
  double* acc = (double*)(ws + 786432 + 1048576 + 2048);       // 16 B

  prep_kernel<<<BSZ, 256, 0, stream>>>(emb, e16, sqn, acc);
  gemm_dist_kernel<<<256, 128, 0, stream>>>(e16, sqn, dst);
  triplet_kernel<<<BSZ, 256, 0, stream>>>(types, dst, acc);
  finalize_kernel<<<1, 1, 0, stream>>>(acc, (float*)d_out);
}
